// SelfAttention_1554778161532
// MI455X (gfx1250) — matrix operand order repaired and checked by reading, not compiled, not run on hardware
//
#include <hip/hip_runtime.h>
#include <math.h>

// ---------------------------------------------------------------------------
// Self-attention block (GroupNorm -> QKV -> softmax attn -> proj + residual)
// B=4, H=W=64, C=256, N=H*W=4096, GROUPS=8, eps=1e-3, scale = C^-0.5 = 1/16
//
// MI455X plan: compute-bound (~77 GFLOP vs ~130MB HBM traffic at 23.3TB/s),
// so all GEMMs run on v_wmma_f32_16x16x32_bf16 (bf16 in, f32 accumulate).
// Attention is flash-style (online softmax); K/V tiles are staged into LDS
// once per workgroup with gfx1250 async global->LDS copies (ASYNCcnt) so all
// 8 waves share one copy instead of 8x redundant VMEM traffic.
// ---------------------------------------------------------------------------

typedef __attribute__((ext_vector_type(16))) __bf16 v16bf;
typedef __attribute__((ext_vector_type(8)))  __bf16 v8bf;
typedef __attribute__((ext_vector_type(8)))  float  v8f;

union BF16x16 { v16bf v; v8bf h[2]; };

#define NPIX   4096      // H*W
#define CDIM   256
#define BATCH  4
#define MROWS  (BATCH * NPIX)   // 16384

// Async 16B global->LDS copy (per lane). VDST = LDS byte address; per the ISA
// aperture rule the low 32 bits of a flat LDS pointer ARE the LDS offset.
__device__ __forceinline__ void async_copy_b128(const __bf16* gsrc, __bf16* ldst) {
  const unsigned loff = (unsigned)(size_t)ldst;
  const unsigned long long gaddr = (unsigned long long)gsrc;
  asm volatile("global_load_async_to_lds_b128 %0, %1, off"
               :: "v"(loff), "v"(gaddr) : "memory");
}

__device__ __forceinline__ void wait_async0() {
  asm volatile("s_wait_asynccnt 0x0" ::: "memory");
}

// ---------------------------------------------------------------------------
// GroupNorm stats: one block per (batch, group); reduce 4096*32 elements.
// ---------------------------------------------------------------------------
__global__ __launch_bounds__(256) void gn_stats(const float* __restrict__ x,
                                                float* __restrict__ stats) {
  const int b = blockIdx.x >> 3;
  const int g = blockIdx.x & 7;
  const float* xb = x + (size_t)b * NPIX * CDIM + g * 32;
  float s = 0.f, s2 = 0.f;
  for (int idx = threadIdx.x; idx < NPIX * 32; idx += 256) {
    const int p = idx >> 5, ch = idx & 31;
    const float v = xb[(size_t)p * CDIM + ch];
    s += v; s2 += v * v;
  }
  const int lane = threadIdx.x & 31, wid = threadIdx.x >> 5;
  for (int m = 16; m >= 1; m >>= 1) {
    s  += __shfl_xor(s,  m, 32);
    s2 += __shfl_xor(s2, m, 32);
  }
  __shared__ float shs[8], shs2[8];
  if (lane == 0) { shs[wid] = s; shs2[wid] = s2; }
  __syncthreads();
  if (threadIdx.x == 0) {
    float S = 0.f, S2 = 0.f;
    for (int i = 0; i < 8; ++i) { S += shs[i]; S2 += shs2[i]; }
    const float inv_n = 1.0f / (NPIX * 32.0f);
    const float mean = S * inv_n;
    const float var  = S2 * inv_n - mean * mean;
    stats[2 * blockIdx.x]     = mean;
    stats[2 * blockIdx.x + 1] = rsqrtf(var + 1e-3f);
  }
}

// ---------------------------------------------------------------------------
// GroupNorm apply + cast to bf16: xn[b,p,c] bf16
// ---------------------------------------------------------------------------
__global__ __launch_bounds__(256) void gn_apply(const float* __restrict__ x,
                                                const float* __restrict__ gamma,
                                                const float* __restrict__ beta,
                                                const float* __restrict__ stats,
                                                __bf16* __restrict__ xn) {
  const size_t i = (size_t)blockIdx.x * 256 + threadIdx.x;   // < 4M
  const int c = (int)(i & 255);
  const int b = (int)(i >> 20);                              // /(4096*256)
  const int g = c >> 5;
  const float mean = stats[(b * 8 + g) * 2];
  const float rstd = stats[(b * 8 + g) * 2 + 1];
  xn[i] = (__bf16)(((x[i] - mean) * rstd) * gamma[c] + beta[c]);
}

// ---------------------------------------------------------------------------
// Weight prep: Wt[n,k] = bf16(W[k,n])  (256x256)
// ---------------------------------------------------------------------------
__global__ __launch_bounds__(256) void wprep(const float* __restrict__ w,
                                             __bf16* __restrict__ wt) {
  const int i = blockIdx.x * 256 + threadIdx.x;   // < 65536
  const int k = i >> 8, n = i & 255;
  wt[n * 256 + k] = (__bf16)w[k * 256 + n];
}

// ---------------------------------------------------------------------------
// WMMA GEMM: C[M=16384, N=256] = A(bf16, row-major, lda=256) @ Wt^T
//   Wt stored [N][K] bf16 so B fragments load contiguously per lane.
// mode 0: outB[row*256+col]           = bf16(acc + bias[col])      (Q, K)
// mode 1: outB[b*N*C + col*N + j]     = bf16(acc + bias[col])      (V transposed)
// mode 2: outF[row*256+col]           = resid + acc + bias[col]    (final proj)
// Each wave: 16 rows x 64 cols (4 C-tiles). Block = 4 waves = 64 rows.
// grid = (256, 4)
// ---------------------------------------------------------------------------
__global__ __launch_bounds__(128) void wmma_gemm(const __bf16* __restrict__ A,
                                                 const __bf16* __restrict__ Wt,
                                                 const float* __restrict__ bias,
                                                 __bf16* __restrict__ outB,
                                                 float* __restrict__ outF,
                                                 const float* __restrict__ resid,
                                                 int mode) {
  const int lane = threadIdx.x & 31;
  const int wv   = threadIdx.x >> 5;
  const int r0   = (blockIdx.x * 4 + wv) * 16;
  const int n0   = blockIdx.y * 64;

  v8f acc[4] = {};

  // A fragment base: lane<16 -> row lane,    K {0..15}
  //                  lane>=16 -> row lane-16, K {16..31}
  const __bf16* abase = A + (size_t)(r0 + (lane & 15)) * CDIM + ((lane < 16) ? 0 : 16);

  for (int kc = 0; kc < 8; ++kc) {
    const int c0 = kc * 32;
    BF16x16 a;
    a.h[0] = *(const v8bf*)(abase + c0);
    a.h[1] = *(const v8bf*)(abase + c0 + 8);
#pragma unroll
    for (int t = 0; t < 4; ++t) {
      // B fragment: lane<16 -> col lane,     K rows 0..15 (contiguous along K)
      //             lane>=16 -> col lane-16, K rows 16..31
      const __bf16* bbase = Wt + (size_t)(n0 + t * 16 + (lane & 15)) * CDIM
                               + c0 + ((lane < 16) ? 0 : 16);
      BF16x16 bfr;
      bfr.h[0] = *(const v8bf*)(bbase);
      bfr.h[1] = *(const v8bf*)(bbase + 8);
      acc[t] = __builtin_amdgcn_wmma_f32_16x16x32_bf16(
          false, a.v, false, bfr.v, (short)0, acc[t], false, false);
    }
  }

  // Epilogue. C layout: VGPR r -> row r (lanes 0-15) / row r+8 (lanes 16-31),
  // col = lane & 15.
  const int rowoff = (lane < 16) ? 0 : 8;
  const int colL   = lane & 15;
#pragma unroll
  for (int t = 0; t < 4; ++t) {
    const int col = n0 + t * 16 + colL;
    const float bv = bias[col];
#pragma unroll
    for (int r = 0; r < 8; ++r) {
      const int row = r0 + r + rowoff;
      const float v = acc[t][r] + bv;
      if (mode == 0) {
        outB[(size_t)row * CDIM + col] = (__bf16)v;
      } else if (mode == 1) {
        const int b = row >> 12;            // /4096
        const int j = row & (NPIX - 1);
        outB[(size_t)b * NPIX * CDIM + (size_t)col * NPIX + j] = (__bf16)v;
      } else {
        const size_t idx = (size_t)row * CDIM + col;
        outF[idx] = resid[idx] + v;
      }
    }
  }
}

// ---------------------------------------------------------------------------
// Fused flash attention. grid = (NPIX/128, BATCH), block = 256 (8 waves).
// Each wave owns 16 query rows, full 16x256 f32 accumulator (16 WMMA tiles),
// streams keys/values 32 at a time with online softmax. K/V tiles staged in
// LDS via async global->LDS copies, shared by all 8 waves.
//   q,k: [b][n][c] bf16 ;  vt: [b][c][n] bf16 ;  o: [b][n][c] bf16
// ---------------------------------------------------------------------------
#define KT_LD 264   // 256 + 8 pad: 528B row stride, 16B aligned, bank-spread
#define VT_LD 40    // 32 + 8 pad:   80B row stride, 16B aligned, bank-spread
#define PT_LD 40

__global__ __launch_bounds__(256, 1) void attn_flash(const __bf16* __restrict__ q,
                                                     const __bf16* __restrict__ k,
                                                     const __bf16* __restrict__ vt,
                                                     __bf16* __restrict__ o) {
  const int lane = threadIdx.x & 31;
  const int wid  = threadIdx.x >> 5;                 // 0..7
  const int tid  = threadIdx.x;
  const int b    = blockIdx.y;
  const int i0   = blockIdx.x * 128 + wid * 16;      // first query row of wave

  const __bf16* qb = q  + (size_t)b * NPIX * CDIM;
  const __bf16* kb = k  + (size_t)b * NPIX * CDIM;
  const __bf16* vb = vt + (size_t)b * NPIX * CDIM;

  __shared__ __align__(16) __bf16 kt[32][KT_LD];     // K rows j0..j0+31
  __shared__ __align__(16) __bf16 vtile[256][VT_LD]; // Vt[c][j0..j0+31]
  __shared__ __align__(16) __bf16 plds[8][16][PT_LD];// per-wave P repack tile

  // Preload Q A-fragments for all 8 K-chunks (K=c dimension, 256 = 8*32).
  v16bf qf[8];
  {
    const __bf16* abase = qb + (size_t)(i0 + (lane & 15)) * CDIM + ((lane < 16) ? 0 : 16);
#pragma unroll
    for (int kc = 0; kc < 8; ++kc) {
      BF16x16 t;
      t.h[0] = *(const v8bf*)(abase + kc * 32);
      t.h[1] = *(const v8bf*)(abase + kc * 32 + 8);
      qf[kc] = t.v;
    }
  }

  v8f acc[16] = {};
  float m[8], l[8];
#pragma unroll
  for (int r = 0; r < 8; ++r) { m[r] = -1e30f; l[r] = 0.f; }

  const float scale = 0.0625f;   // 256^-0.5
  const int rowoff = (lane < 16) ? 0 : 8;
  const int colL   = lane & 15;

  for (int j0 = 0; j0 < NPIX; j0 += 32) {
    // ---- async-stage K and V tiles into LDS (all 256 threads) -------------
    __syncthreads();   // previous iteration finished reading LDS tiles
#pragma unroll
    for (int t = 0; t < 4; ++t) {
      const int idx = tid + t * 256;                 // 0..1023
      const int kr = idx >> 5, kcol = (idx & 31) * 8;
      async_copy_b128(kb + (size_t)(j0 + kr) * CDIM + kcol, &kt[kr][kcol]);
    }
#pragma unroll
    for (int t = 0; t < 4; ++t) {
      const int idx = tid + t * 256;                 // 0..1023
      const int c = idx >> 2, part = (idx & 3) * 8;
      async_copy_b128(vb + (size_t)c * NPIX + j0 + part, &vtile[c][part]);
    }
    wait_async0();
    __syncthreads();   // tiles visible to all waves

    // ---- S tiles: 16 queries x 32 keys (K fragments from LDS) -------------
    v8f s0 = {}, s1 = {};
    const __bf16* krow0 = &kt[colL]     [(lane < 16) ? 0 : 16];
    const __bf16* krow1 = &kt[colL + 16][(lane < 16) ? 0 : 16];
#pragma unroll
    for (int kc = 0; kc < 8; ++kc) {
      BF16x16 b0, b1;
      b0.h[0] = *(const v8bf*)(krow0 + kc * 32);
      b0.h[1] = *(const v8bf*)(krow0 + kc * 32 + 8);
      b1.h[0] = *(const v8bf*)(krow1 + kc * 32);
      b1.h[1] = *(const v8bf*)(krow1 + kc * 32 + 8);
      s0 = __builtin_amdgcn_wmma_f32_16x16x32_bf16(false, qf[kc], false, b0.v,
                                                   (short)0, s0, false, false);
      s1 = __builtin_amdgcn_wmma_f32_16x16x32_bf16(false, qf[kc], false, b1.v,
                                                   (short)0, s1, false, false);
    }

    // ---- online softmax update --------------------------------------------
    float corr[8];
#pragma unroll
    for (int r = 0; r < 8; ++r) {
      float t = fmaxf(s0[r], s1[r]) * scale;
      t = fmaxf(t, __shfl_xor(t, 1, 32));
      t = fmaxf(t, __shfl_xor(t, 2, 32));
      t = fmaxf(t, __shfl_xor(t, 4, 32));
      t = fmaxf(t, __shfl_xor(t, 8, 32));       // row max (within 16-lane half)
      const float mn = fmaxf(m[r], t);
      corr[r] = __expf(m[r] - mn);
      m[r] = mn;
      const float p0 = __expf(s0[r] * scale - mn);
      const float p1 = __expf(s1[r] * scale - mn);
      s0[r] = p0; s1[r] = p1;
      float rs = p0 + p1;
      rs += __shfl_xor(rs, 1, 32);
      rs += __shfl_xor(rs, 2, 32);
      rs += __shfl_xor(rs, 4, 32);
      rs += __shfl_xor(rs, 8, 32);
      l[r] = l[r] * corr[r] + rs;
    }
#pragma unroll
    for (int t = 0; t < 16; ++t)
#pragma unroll
      for (int r = 0; r < 8; ++r)
        acc[t][r] *= corr[r];

    // ---- repack P (C layout) into A-fragment layout via per-wave LDS ------
#pragma unroll
    for (int r = 0; r < 8; ++r) {
      plds[wid][r + rowoff][colL]      = (__bf16)s0[r];
      plds[wid][r + rowoff][colL + 16] = (__bf16)s1[r];
    }
    asm volatile("s_wait_dscnt 0" ::: "memory");   // same-wave LDS RAW fence
    v16bf pf;
    {
      const __bf16* pb = &plds[wid][colL][(lane < 16) ? 0 : 16];
      BF16x16 t;
      t.h[0] = *(const v8bf*)(pb);
      t.h[1] = *(const v8bf*)(pb + 8);
      pf = t.v;
    }
    asm volatile("s_wait_dscnt 0" ::: "memory");   // reads done before overwrite

    // ---- acc += P @ V  (V fragments from LDS) -----------------------------
#pragma unroll
    for (int nt = 0; nt < 16; ++nt) {
      const __bf16* vrow = &vtile[nt * 16 + colL][(lane < 16) ? 0 : 16];
      BF16x16 bv;
      bv.h[0] = *(const v8bf*)(vrow);
      bv.h[1] = *(const v8bf*)(vrow + 8);
      acc[nt] = __builtin_amdgcn_wmma_f32_16x16x32_bf16(
          false, pf, false, bv.v, (short)0, acc[nt], false, false);
    }
  }

  // ---- normalize and store ------------------------------------------------
  __bf16* ob = o + (size_t)b * NPIX * CDIM;
  float inv[8];
#pragma unroll
  for (int r = 0; r < 8; ++r) inv[r] = 1.0f / l[r];
#pragma unroll
  for (int nt = 0; nt < 16; ++nt)
#pragma unroll
    for (int r = 0; r < 8; ++r)
      ob[(size_t)(i0 + r + rowoff) * CDIM + nt * 16 + colL] =
          (__bf16)(acc[nt][r] * inv[r]);
}

// ---------------------------------------------------------------------------
// Launch
// ---------------------------------------------------------------------------
extern "C" void kernel_launch(void* const* d_in, const int* in_sizes, int n_in,
                              void* d_out, int out_size, void* d_ws, size_t ws_size,
                              hipStream_t stream) {
  (void)in_sizes; (void)n_in; (void)out_size; (void)ws_size;
  const float* x     = (const float*)d_in[0];
  const float* gamma = (const float*)d_in[1];
  const float* beta  = (const float*)d_in[2];
  const float* wq    = (const float*)d_in[3];
  const float* bq    = (const float*)d_in[4];
  const float* wk    = (const float*)d_in[5];
  const float* bk    = (const float*)d_in[6];
  const float* wv    = (const float*)d_in[7];
  const float* bv    = (const float*)d_in[8];
  const float* wp    = (const float*)d_in[9];
  const float* bp    = (const float*)d_in[10];
  float* out = (float*)d_out;

  char* ws = (char*)d_ws;
  size_t off = 0;
  auto alloc = [&](size_t bytes) -> void* {
    void* p = ws + off;
    off += (bytes + 255) & ~(size_t)255;
    return p;
  };
  const size_t act_bytes = (size_t)MROWS * CDIM * sizeof(__bf16);   // 8 MB
  __bf16* xn  = (__bf16*)alloc(act_bytes);
  __bf16* qb  = (__bf16*)alloc(act_bytes);
  __bf16* kbf = (__bf16*)alloc(act_bytes);
  __bf16* vtb = (__bf16*)alloc(act_bytes);
  __bf16* ob  = (__bf16*)alloc(act_bytes);
  __bf16* wqt = (__bf16*)alloc(CDIM * CDIM * sizeof(__bf16));
  __bf16* wkt = (__bf16*)alloc(CDIM * CDIM * sizeof(__bf16));
  __bf16* wvt = (__bf16*)alloc(CDIM * CDIM * sizeof(__bf16));
  __bf16* wpt = (__bf16*)alloc(CDIM * CDIM * sizeof(__bf16));
  float*  stats = (float*)alloc(BATCH * 8 * 2 * sizeof(float));

  // GroupNorm
  gn_stats<<<BATCH * 8, 256, 0, stream>>>(x, stats);
  gn_apply<<<(MROWS * CDIM) / 256, 256, 0, stream>>>(x, gamma, beta, stats, xn);

  // Weight transposes (bf16)
  wprep<<<CDIM * CDIM / 256, 256, 0, stream>>>(wq, wqt);
  wprep<<<CDIM * CDIM / 256, 256, 0, stream>>>(wk, wkt);
  wprep<<<CDIM * CDIM / 256, 256, 0, stream>>>(wv, wvt);
  wprep<<<CDIM * CDIM / 256, 256, 0, stream>>>(wp, wpt);

  // Q, K, V projections (V stored transposed per batch)
  dim3 ggrid(MROWS / 64, CDIM / 64);
  wmma_gemm<<<ggrid, 128, 0, stream>>>(xn, wqt, bq, qb,  nullptr, nullptr, 0);
  wmma_gemm<<<ggrid, 128, 0, stream>>>(xn, wkt, bk, kbf, nullptr, nullptr, 0);
  wmma_gemm<<<ggrid, 128, 0, stream>>>(xn, wvt, bv, vtb, nullptr, nullptr, 1);

  // Fused flash attention
  attn_flash<<<dim3(NPIX / 128, BATCH), 256, 0, stream>>>(qb, kbf, vtb, ob);

  // Output projection + bias + residual (f32)
  wmma_gemm<<<ggrid, 128, 0, stream>>>(ob, wpt, bp, nullptr, out, x, 2);
}
